// MultiHeadSelfAttention_24481313587606
// MI455X (gfx1250) — matrix operand order repaired and checked by reading, not compiled, not run on hardware
//
#include <hip/hip_runtime.h>
#include <hip/hip_bf16.h>

typedef __bf16 bf16_t;
typedef __attribute__((ext_vector_type(16))) __bf16 v16bf;
typedef __attribute__((ext_vector_type(8)))  __bf16 v8bf;
typedef __attribute__((ext_vector_type(8)))  float  v8f;

#define WMMA_BF16(a, b, c) \
  __builtin_amdgcn_wmma_f32_16x16x32_bf16(false, (a), false, (b), (short)0, (c), false, false)

// ---- CDNA5 Tensor Data Mover support (guarded; falls back to manual loads) --
#if defined(__has_builtin)
#if __has_builtin(__builtin_amdgcn_tensor_load_to_lds)
#define HAVE_TDM 1
#if defined(__clang_major__) && (__clang_major__ >= 23)
#pragma message("CDNA5: tensor_load_to_lds builtin PRESENT, using 6-arg form")
#else
#pragma message("CDNA5: tensor_load_to_lds builtin PRESENT, using 5-arg form")
#endif
#else
#pragma message("CDNA5: tensor_load_to_lds builtin ABSENT, using fallback loads")
#endif
#endif
#ifndef HAVE_TDM
#define HAVE_TDM 0
#endif

#if HAVE_TDM
typedef __attribute__((ext_vector_type(4))) unsigned int u32x4;
typedef __attribute__((ext_vector_type(8))) int          i32x8;
typedef __attribute__((ext_vector_type(4))) int          i32x4;

// Issue one TDM copy of a 64x64 bf16 tile (row stride 64 elems in memory) into
// LDS with a padded row stride of 72 bf16 (pad 4 DWORDs every 32 DWORDs).
// D# layout per CDNA5 ISA.
__device__ __forceinline__ void tdm_load_tile_64x64(const bf16_t* gsrc,
                                                    unsigned lds_off) {
  unsigned long long ga = (unsigned long long)(uintptr_t)gsrc;
  u32x4 g0;
  g0[0] = 1u;                                        // count=1 (valid), user mode
  g0[1] = lds_off;                                   // lds_addr (bytes)
  g0[2] = (unsigned)(ga & 0xffffffffu);              // global_addr[31:0]
  g0[3] = (unsigned)((ga >> 32) & 0x1ffffffu)        // global_addr[56:32]
        | (2u << 30);                                // type = 2 ("image")
  i32x8 g1;
  g1[0] = (1 << 16)                                  // data_size = 2 bytes
        | (1 << 20)                                  // pad_enable
        | (4 << 22)                                  // pad_interval: 32 DWORDs
        | (3 << 25);                                 // pad_amount:   4 DWORDs
  g1[1] = (64 << 16);                                // tensor_dim0 = 64 (lo16)
  g1[2] = (0x1000 << 16);                            // tensor_dim1 = 4096 (lo16)
  g1[3] = (64 << 16);                                // tile_dim0 = 64
  g1[4] = 64;                                        // tile_dim1 = 64 (tile_dim2=0)
  g1[5] = 64;                                        // tensor_dim0_stride = 64
  g1[6] = 0;
  g1[7] = 0;
  i32x4 z4 = {0, 0, 0, 0};
#if defined(__clang_major__) && (__clang_major__ >= 23)
  i32x8 z8 = {0, 0, 0, 0, 0, 0, 0, 0};
  __builtin_amdgcn_tensor_load_to_lds(g0, g1, z4, z4, z8, 0);
#else
  __builtin_amdgcn_tensor_load_to_lds(g0, g1, z4, z4, 0);
#endif
}
#endif  // HAVE_TDM

__device__ __forceinline__ void wait_tensor0() {
#if defined(__has_builtin) && __has_builtin(__builtin_amdgcn_s_wait_tensorcnt)
  __builtin_amdgcn_s_wait_tensorcnt(0);
#else
  asm volatile("s_wait_tensorcnt 0x0" ::: "memory");
#endif
}

// v_exp_f32 is natively exp2: keep the whole softmax in log2 domain
// (log2(e)/sqrt(hd) is pre-folded into W_q).
__device__ __forceinline__ float fast_exp2(float x) {
#if defined(__has_builtin) && __has_builtin(__builtin_amdgcn_exp2f)
  return __builtin_amdgcn_exp2f(x);
#else
  return exp2f(x);
#endif
}

__device__ __forceinline__ float fast_rcp(float x) {
#if defined(__has_builtin) && __has_builtin(__builtin_amdgcn_rcpf)
  return __builtin_amdgcn_rcpf(x);
#else
  return 1.0f / x;
#endif
}

// ---------------------------------------------------------------------------
// Fragment loaders following the CDNA5 WMMA VGPR layouts (CDNA5 ISA §7.12.2)
// ---------------------------------------------------------------------------

// A fragment (16x32 bf16, MxK): lane L<16 -> row L, K = {0..7, 16..23};
// lane L>=16 -> row L-16, K = {8..15, 24..31}.  src row-major, ld elements.
__device__ __forceinline__ v16bf load_a_frag(const bf16_t* src, int ld,
                                             int row0, int k0, int lane) {
  int row = row0 + (lane & 15);
  int kb  = (lane >> 4) * 8;
  const bf16_t* p = src + (size_t)row * ld + k0 + kb;
  v8bf lo = *(const v8bf*)(p);        // K = kb .. kb+7
  v8bf hi = *(const v8bf*)(p + 16);   // K = kb+16 .. kb+23
  v16bf a;
#pragma unroll
  for (int i = 0; i < 8; ++i) { a[i] = lo[i]; a[i + 8] = hi[i]; }
  return a;
}

// B fragment (32x16 bf16, KxN) where B[k][n] = src[(col0+n)*ld + k0 + k]
// ("NT" access: contraction contiguous in memory). lanes 0-15: K = {0..7, 16..23},
// lanes 16-31: K = {8..15, 24..31}; lane column = lane&15.
__device__ __forceinline__ v16bf load_b_frag(const bf16_t* src, int ld,
                                             int col0, int k0, int lane) {
  int col = col0 + (lane & 15);
  int kb  = (lane >> 4) * 8;
  const bf16_t* p = src + (size_t)col * ld + k0 + kb;
  v8bf lo = *(const v8bf*)(p);
  v8bf hi = *(const v8bf*)(p + 16);
  v16bf b;
#pragma unroll
  for (int i = 0; i < 8; ++i) { b[i] = lo[i]; b[i + 8] = hi[i]; }
  return b;
}

// ---------------------------------------------------------------------------
// 0) fp32 -> bf16 convert (with optional scale)
// ---------------------------------------------------------------------------
__global__ void convert_bf16_kernel(const float* __restrict__ src,
                                    bf16_t* __restrict__ dst, int n, float scale) {
  int i = blockIdx.x * blockDim.x + threadIdx.x;
  if (i < n) dst[i] = (bf16_t)(src[i] * scale);
}

// ---------------------------------------------------------------------------
// 1) QKV projection GEMM:  y[m,n] = sum_k x[m,k] * W[n,k]
//    M=8192, N=512, K=512.  Output stored head-major: dst[((b*8+h)*4096+t)*64+d]
//    Block = 256 threads (8 waves): 2 waves in M x 4 waves in N -> 32x256 tile.
// ---------------------------------------------------------------------------
__global__ __launch_bounds__(256) void qkv_kernel(
    const bf16_t* __restrict__ xb,
    const bf16_t* __restrict__ Wq, const bf16_t* __restrict__ Wk,
    const bf16_t* __restrict__ Wv,
    bf16_t* __restrict__ Qb, bf16_t* __restrict__ Kb, bf16_t* __restrict__ Vb) {
  const bf16_t* W;
  bf16_t* dst;
  if (blockIdx.z == 0)      { W = Wq; dst = Qb; }
  else if (blockIdx.z == 1) { W = Wk; dst = Kb; }
  else                      { W = Wv; dst = Vb; }

  int lane = threadIdx.x & 31, wid = threadIdx.x >> 5;
  int wm = wid & 1, wn = wid >> 1;
  int m0 = blockIdx.y * 32 + wm * 16;
  int n0 = blockIdx.x * 256 + wn * 64;

  v8f acc[4] = {};
  for (int k0 = 0; k0 < 512; k0 += 32) {
    v16bf a = load_a_frag(xb, 512, m0, k0, lane);
#pragma unroll
    for (int s = 0; s < 4; ++s) {
      v16bf b = load_b_frag(W, 512, n0 + 16 * s, k0, lane);
      acc[s] = WMMA_BF16(a, b, acc[s]);
    }
  }

  int lane16 = lane & 15, hi = lane >> 4;
#pragma unroll
  for (int s = 0; s < 4; ++s) {
    int col = n0 + 16 * s + lane16;
    int h = col >> 6, d = col & 63;
#pragma unroll
    for (int r = 0; r < 8; ++r) {
      int m = m0 + r + 8 * hi;
      int bb = m >> 12, t = m & 4095;
      dst[((size_t)(bb * 8 + h) * 4096 + t) * 64 + d] = (bf16_t)acc[s][r];
    }
  }
}

// ---------------------------------------------------------------------------
// 2) Flash attention. Per block: 128 query rows of one (b,h); 8 waves each own
//    16 rows. K tile pulled by the Tensor Data Mover directly into padded LDS;
//    V transposed into LDS via registers. Scores never leave registers.
//    log2(e)/sqrt(64) pre-folded into Q; softmax runs in exp2 domain.
// ---------------------------------------------------------------------------
#define LDK 72  // padded LDS row stride (keeps 16B alignment: 72*2=144=9*16)

__global__ __launch_bounds__(256) void attn_kernel(
    const bf16_t* __restrict__ Qb, const bf16_t* __restrict__ Kb,
    const bf16_t* __restrict__ Vb, bf16_t* __restrict__ ctxb) {
  __shared__ bf16_t Kl[64 * LDK];       // [key][k]   row-major (TDM-padded)
  __shared__ bf16_t Vt[64 * LDK];       // [dim][key] transposed
  __shared__ bf16_t Pl[8 * 16 * LDK];   // per-wave P tile [16][LDK]

  int bh = blockIdx.y;
  int bb = bh >> 3, h = bh & 7;
  int lane = threadIdx.x & 31, wid = threadIdx.x >> 5;
  int lane16 = lane & 15, hi = lane >> 4;
  int q0 = blockIdx.x * 128 + wid * 16;

  const bf16_t* Qh = Qb + (size_t)bh * 4096 * 64;
  const bf16_t* Kh = Kb + (size_t)bh * 4096 * 64;
  const bf16_t* Vh = Vb + (size_t)bh * 4096 * 64;

  v16bf qfrag0 = load_a_frag(Qh, 64, q0, 0, lane);
  v16bf qfrag1 = load_a_frag(Qh, 64, q0, 32, lane);

  v8f o[4] = {};
  float mrow[8], lrow[8];
#pragma unroll
  for (int r = 0; r < 8; ++r) { mrow[r] = -__builtin_inff(); lrow[r] = 0.f; }

  bf16_t* Pw = Pl + wid * 16 * LDK;
#if HAVE_TDM
  unsigned ldsKoff = (unsigned)(uintptr_t)(&Kl[0]);
#endif

  for (int kc = 0; kc < 4096; kc += 64) {
    __syncthreads();   // previous iteration's consumers are done with Kl/Vt

#if HAVE_TDM
    if (wid == 0) tdm_load_tile_64x64(Kh + (size_t)kc * 64, ldsKoff);
#endif
    // V tile: registers -> transposed LDS store. (K fallback path if no TDM.)
    {
      int t   = threadIdx.x;
      int key = t >> 2;            // 0..63
      int seg = (t & 3) * 16;      // 0,16,32,48
#if !HAVE_TDM
      const bf16_t* kp = Kh + (size_t)(kc + key) * 64 + seg;
      v8bf ka = *(const v8bf*)(kp);
      v8bf kb = *(const v8bf*)(kp + 8);
      *(v8bf*)(Kl + key * LDK + seg)     = ka;
      *(v8bf*)(Kl + key * LDK + seg + 8) = kb;
      if (kc + 64 < 4096) __builtin_prefetch(kp + 64 * 64, 0, 1);
#endif
      const bf16_t* vp = Vh + (size_t)(kc + key) * 64 + seg;
      v8bf va = *(const v8bf*)(vp);
      v8bf vb = *(const v8bf*)(vp + 8);
      if (kc + 64 < 4096) __builtin_prefetch(vp + 64 * 64, 0, 1);  // next chunk
#pragma unroll
      for (int i = 0; i < 8; ++i) {
        Vt[(seg + i) * LDK + key]     = va[i];   // transpose during store
        Vt[(seg + 8 + i) * LDK + key] = vb[i];
      }
    }
#if HAVE_TDM
    if (wid == 0) wait_tensor0();  // TENSORcnt -> 0 before the barrier
#endif
    __syncthreads();

    // S = Q K^T  (16 q-rows x 64 keys, fp32 accum; already in log2 domain)
    v8f S[4];
#pragma unroll
    for (int s = 0; s < 4; ++s) {
      v16bf b0 = load_b_frag(Kl, LDK, 16 * s, 0, lane);
      v16bf b1 = load_b_frag(Kl, LDK, 16 * s, 32, lane);
      v8f z = {};
      z = WMMA_BF16(qfrag0, b0, z);
      S[s] = WMMA_BF16(qfrag1, b1, z);
    }

    // Online softmax (exp2 domain); row stats reduced across 16-lane halves.
#pragma unroll
    for (int r = 0; r < 8; ++r) {
      float v = fmaxf(fmaxf(S[0][r], S[1][r]), fmaxf(S[2][r], S[3][r]));
#pragma unroll
      for (int msk = 1; msk < 16; msk <<= 1) v = fmaxf(v, __shfl_xor(v, msk, 32));
      float mn    = fmaxf(mrow[r], v);
      float alpha = fast_exp2(mrow[r] - mn);
      mrow[r] = mn;
      lrow[r] *= alpha;
#pragma unroll
      for (int s = 0; s < 4; ++s) o[s][r] *= alpha;
      float psum = 0.f;
#pragma unroll
      for (int s = 0; s < 4; ++s) {
        float p = fast_exp2(S[s][r] - mn);
        S[s][r] = p;
        psum += p;
      }
#pragma unroll
      for (int msk = 1; msk < 16; msk <<= 1) psum += __shfl_xor(psum, msk, 32);
      lrow[r] += psum;
    }

    // P -> per-wave LDS (C layout -> row-major), then consume as A fragments.
#pragma unroll
    for (int s = 0; s < 4; ++s)
#pragma unroll
      for (int r = 0; r < 8; ++r)
        Pw[(r + 8 * hi) * LDK + 16 * s + lane16] = (bf16_t)S[s][r];

    // O += P V   (contraction over 64 keys)
#pragma unroll
    for (int ks = 0; ks < 2; ++ks) {
      v16bf a = load_a_frag(Pw, LDK, 0, 32 * ks, lane);
#pragma unroll
      for (int s = 0; s < 4; ++s) {
        v16bf bv = load_b_frag(Vt, LDK, 16 * s, 32 * ks, lane);
        o[s] = WMMA_BF16(a, bv, o[s]);
      }
    }
  }

  // ctx (B,T,D) bf16 for the output projection
  float inv[8];
#pragma unroll
  for (int r = 0; r < 8; ++r) inv[r] = fast_rcp(lrow[r]);
#pragma unroll
  for (int s = 0; s < 4; ++s)
#pragma unroll
    for (int r = 0; r < 8; ++r) {
      int t = q0 + r + 8 * hi;
      float v = o[s][r] * inv[r];
      ctxb[((size_t)(bb * 4096 + t)) * 512 + h * 64 + 16 * s + lane16] = (bf16_t)v;
    }
}

// ---------------------------------------------------------------------------
// 3) Output projection + residual + LayerNorm, fused.
//    Block = 16 rows; 8 waves split N=512 (64 cols each). LN reduction:
//    shfl within 16-lane halves, then cross-wave via LDS.
// ---------------------------------------------------------------------------
__global__ __launch_bounds__(256) void proj_ln_kernel(
    const bf16_t* __restrict__ ctxb, const bf16_t* __restrict__ Wob,
    const float* __restrict__ x, const float* __restrict__ gamma,
    const float* __restrict__ beta, float* __restrict__ out) {
  __shared__ float red[16][8][2];
  int lane = threadIdx.x & 31, wid = threadIdx.x >> 5;
  int lane16 = lane & 15, hi = lane >> 4;
  int m0 = blockIdx.x * 16;

  v8f acc[4] = {};
  for (int k0 = 0; k0 < 512; k0 += 32) {
    v16bf a = load_a_frag(ctxb, 512, m0, k0, lane);
#pragma unroll
    for (int s = 0; s < 4; ++s) {
      v16bf b = load_b_frag(Wob, 512, wid * 64 + 16 * s, k0, lane);
      acc[s] = WMMA_BF16(a, b, acc[s]);
    }
  }

  float y[4][8];
  float psum[8], psq[8];
#pragma unroll
  for (int r = 0; r < 8; ++r) {
    int m = m0 + r + 8 * hi;
    float s1 = 0.f, s2 = 0.f;
#pragma unroll
    for (int s = 0; s < 4; ++s) {
      int col = wid * 64 + 16 * s + lane16;
      float v = acc[s][r] + x[(size_t)m * 512 + col];  // residual, fp32
      y[s][r] = v;
      s1 += v;
      s2 += v * v;
    }
#pragma unroll
    for (int msk = 1; msk < 16; msk <<= 1) {
      s1 += __shfl_xor(s1, msk, 32);
      s2 += __shfl_xor(s2, msk, 32);
    }
    psum[r] = s1;
    psq[r]  = s2;
  }
  if (lane16 == 0) {
#pragma unroll
    for (int r = 0; r < 8; ++r) {
      red[r + 8 * hi][wid][0] = psum[r];
      red[r + 8 * hi][wid][1] = psq[r];
    }
  }
  __syncthreads();
#pragma unroll
  for (int r = 0; r < 8; ++r) {
    int row = r + 8 * hi;
    float s1 = 0.f, s2 = 0.f;
#pragma unroll
    for (int w = 0; w < 8; ++w) { s1 += red[row][w][0]; s2 += red[row][w][1]; }
    float mean = s1 * (1.0f / 512.0f);
    float var  = s2 * (1.0f / 512.0f) - mean * mean;
    float rs   = rsqrtf(var + 1e-5f);
    int m = m0 + row;
#pragma unroll
    for (int s = 0; s < 4; ++s) {
      int col = wid * 64 + 16 * s + lane16;
      out[(size_t)m * 512 + col] = (y[s][r] - mean) * rs * gamma[col] + beta[col];
    }
  }
}

// ---------------------------------------------------------------------------
// launch
// ---------------------------------------------------------------------------
extern "C" void kernel_launch(void* const* d_in, const int* in_sizes, int n_in,
                              void* d_out, int out_size, void* d_ws, size_t ws_size,
                              hipStream_t stream) {
  const float* x     = (const float*)d_in[0];
  const float* Wq    = (const float*)d_in[1];
  const float* Wk    = (const float*)d_in[2];
  const float* Wv    = (const float*)d_in[3];
  const float* Wo    = (const float*)d_in[4];
  const float* gamma = (const float*)d_in[5];
  const float* beta  = (const float*)d_in[6];
  float* out = (float*)d_out;

  const int M = 2 * 4096;        // 8192 rows
  const int D = 512;
  const int NX = M * D;          // 4,194,304
  const int NW = D * D;          // 262,144

  bf16_t* xb   = (bf16_t*)d_ws;
  bf16_t* Wqb  = xb + (size_t)NX;
  bf16_t* Wkb  = Wqb + NW;
  bf16_t* Wvb  = Wkb + NW;
  bf16_t* Wob  = Wvb + NW;
  bf16_t* Qb   = Wob + NW;              // (B,H,T,64)
  bf16_t* Kb   = Qb + (size_t)NX;
  bf16_t* Vb   = Kb + (size_t)NX;
  bf16_t* ctxb = Vb + (size_t)NX;       // (B,T,D)

  // fp32 -> bf16; fold log2(e)/sqrt(head_dim) into W_q so softmax can run in
  // the exp2 domain (v_exp_f32 is natively exp2).
  const float qscale = 0.125f * 1.44269504088896340736f;
  convert_bf16_kernel<<<(NX + 255) / 256, 256, 0, stream>>>(x, xb, NX, 1.0f);
  convert_bf16_kernel<<<(NW + 255) / 256, 256, 0, stream>>>(Wq, Wqb, NW, qscale);
  convert_bf16_kernel<<<(NW + 255) / 256, 256, 0, stream>>>(Wk, Wkb, NW, 1.0f);
  convert_bf16_kernel<<<(NW + 255) / 256, 256, 0, stream>>>(Wv, Wvb, NW, 1.0f);
  convert_bf16_kernel<<<(NW + 255) / 256, 256, 0, stream>>>(Wo, Wob, NW, 1.0f);

  // QKV projections: grid (N/256, M/32, 3)
  qkv_kernel<<<dim3(2, 256, 3), 256, 0, stream>>>(xb, Wqb, Wkb, Wvb, Qb, Kb, Vb);

  // Flash attention: grid (T/128, B*H)
  attn_kernel<<<dim3(32, 16), 256, 0, stream>>>(Qb, Kb, Vb, ctxb);

  // Output projection + residual + LayerNorm: grid (M/16)
  proj_ln_kernel<<<dim3(M / 16), 256, 0, stream>>>(ctxb, Wob, x, gamma, beta, out);
}
